// MwAN_full_66022237274304
// MI455X (gfx1250) — matrix operand order repaired and checked by reading, not compiled, not run on hardware
//
#include <hip/hip_runtime.h>
#include <cstdint>

// ---------------------------------------------------------------------------
// MwAN forward for MI455X (gfx1250, wave32).  All GEMM operands are padded to
// full 16x16x32 WMMA tiles and stored K-major, so k_gemm has exactly one
// branch-free path: paired global_load_b128 per fragment ->
// v_wmma_f32_16x16x32_f16 -> unguarded store.  GRU scans are fused
// persistent-workgroup kernels whose per-step gi slab is staged into LDS by
// the Tensor Data Mover (tensor_load_to_lds + s_wait_tensorcnt) overlapped
// with the h@Whh^T WMMA phase.
// ---------------------------------------------------------------------------

typedef _Float16 f16;
typedef __attribute__((ext_vector_type(16))) _Float16 v16h;
typedef __attribute__((ext_vector_type(8)))  float    v8f;

#if defined(__HIP_DEVICE_COMPILE__) && defined(__gfx1250__)
#define USE_TDM 1
typedef __attribute__((ext_vector_type(4))) unsigned int u32x4;
typedef __attribute__((ext_vector_type(8))) int i32x8;
typedef __attribute__((ext_vector_type(4))) int i32x4;
#endif

#define DEVI __device__ __forceinline__

DEVI float sigm(float x) { return 1.0f / (1.0f + __expf(-x)); }
DEVI float actf(float v, int act) {
  if (act == 1) return sigm(v);
  if (act == 2) return tanhf(v);
  if (act == 3) return v > 0.0f ? v : 0.01f * v;   // leaky_relu
  return v;
}
DEVI float wave_sum(float v) {
#pragma unroll
  for (int o = 16; o; o >>= 1) v += __shfl_xor(v, o, 32);
  return v;
}
DEVI float wave_max(float v) {
#pragma unroll
  for (int o = 16; o; o >>= 1) v = fmaxf(v, __shfl_xor(v, o, 32));
  return v;
}

// ---------------------------------------------------------------------------
// Branch-free WMMA GEMM on padded K-major operands:  C = act( A x B + bias ).
// Requires: gridDim = (N/16, M/16, batch), K % 32 == 0, lda/ldb even.
// A: [M, lda] row-major;  Bw: [N, ldb] row-major over K  (C = A @ Bw^T).
// ---------------------------------------------------------------------------
__global__ __launch_bounds__(32) void k_gemm(
    const f16* __restrict__ A, const f16* __restrict__ Bw,
    const float* __restrict__ bias, float* __restrict__ C, f16* __restrict__ C16,
    int K, int lda, int ldb, int ldc, int act,
    long long sA, long long sB, long long sC)
{
  const int lane = threadIdx.x;
  const int n0 = blockIdx.x * 16;
  const int m0 = blockIdx.y * 16;
  const long long bz = blockIdx.z;
  A += bz * sA; Bw += bz * sB; C += bz * sC;
  if (C16) C16 += bz * sC;

  const int mrow = m0 + (lane & 15);
  const int half = lane >> 4;
  const int ncol = n0 + (lane & 15);
  const f16* arow = A + (long long)mrow * lda;
  const f16* brow = Bw + (long long)ncol * ldb;

  v8f acc = {};
  for (int k0 = 0; k0 < K; k0 += 32) {
    // A fragment: two contiguous 8-half runs per lane (documented A layout)
    union { v16h v; unsigned u[8]; } ua;
    const unsigned* ap = (const unsigned*)(arow + k0);
#pragma unroll
    for (int d = 0; d < 4; ++d) {
      ua.u[d]     = ap[half * 4 + d];
      ua.u[4 + d] = ap[8 + half * 4 + d];
    }
    // B fragment: two contiguous 8-half runs per lane (same layout as A)
    union { v16h v; unsigned u[8]; } ub;
    const unsigned* bp = (const unsigned*)(brow + k0);
#pragma unroll
    for (int d = 0; d < 4; ++d) {
      ub.u[d]     = bp[half * 4 + d];
      ub.u[4 + d] = bp[8 + half * 4 + d];
    }
    if (k0 + 32 < K) __builtin_prefetch(arow + k0 + 32, 0, 1);
    acc = __builtin_amdgcn_wmma_f32_16x16x32_f16(false, ua.v, false, ub.v,
                                                 (short)0, acc, false, false);
  }
  float bv = bias ? bias[ncol] : 0.0f;
#pragma unroll
  for (int r = 0; r < 8; ++r) {
    int m = m0 + half * 8 + r;
    float v = actf(acc[r] + bv, act);
    C[(long long)m * ldc + ncol] = v;
    if (C16) C16[(long long)m * ldc + ncol] = (f16)v;
  }
}

// ---------------------------------------------------------------------------
// Fused one-direction GRU scan.  gi = x@W_ih^T + b_ih precomputed, padded to
// [Bb*T, N3pad].  Whh zero-padded to [N3pad][hidPad] -> WMMA loop branch-free.
// Per-step gi slab (16 rows x N3pad, row stride T*N3pad) is DMA'd into LDS by
// the TDM while the WMMA phase runs; gates read it from LDS.
// ---------------------------------------------------------------------------
#define GRU_HPAD 160   // >= max hidPad (160)
#define GRU_NPAD 464   // >= max N3pad (464)

__global__ __launch_bounds__(256) void k_gru_dir(
    const float* __restrict__ gi, const f16* __restrict__ Whh,
    const float* __restrict__ bhh, float* __restrict__ out,
    int Bb, int T, int hid, int hidPad, int N3pad, int rev,
    int outStride, int outOff)
{
  __shared__ __align__(16) f16   s_h[16 * GRU_HPAD];
  __shared__ __align__(16) float s_gh[16 * GRU_NPAD];
  __shared__ __align__(16) float s_gi[16 * GRU_NPAD];
  const int tid = threadIdx.x;
  const int lane = tid & 31;
  const int wave = tid >> 5;
  const int b0 = blockIdx.x * 16;
  const int N3 = 3 * hid;
  const int ntiles = N3pad >> 4;

  float hreg[10];
#pragma unroll
  for (int i = 0; i < 10; ++i) hreg[i] = 0.0f;
  for (int e = tid; e < 16 * GRU_HPAD; e += 256) s_h[e] = (f16)0.0f;
  __syncthreads();

  for (int t = 0; t < T; ++t) {
    const int tg = rev ? (T - 1 - t) : t;

#ifdef USE_TDM
    // ---- async TDM: stage gi[b0..b0+15, tg, :] (2D tile) into s_gi -------
    if (wave == 0) {
      unsigned long long ga =
          (unsigned long long)(uintptr_t)(gi + ((long long)b0 * T + tg) * N3pad);
      unsigned ldsa = (unsigned)(uintptr_t)(void*)&s_gi[0];
      u32x4 g0;
      g0[0] = 1u;                                     // count=1, user D#
      g0[1] = ldsa;                                   // lds_addr
      g0[2] = (unsigned)(ga & 0xFFFFFFFFu);           // global_addr[31:0]
      g0[3] = (unsigned)(((ga >> 32) & 0x01FFFFFFu) | 0x80000000u); // +type=2
      unsigned dim0 = (unsigned)N3pad;
      unsigned dim1 = (unsigned)(Bb - b0);
      unsigned long long st0 = (unsigned long long)T * (unsigned)N3pad;
      i32x8 g1;
      g1[0] = (int)0x00020000u;                       // data_size=4B
      g1[1] = (int)(dim0 << 16);                      // tensor_dim0 lo
      g1[2] = (int)((dim0 >> 16) | (dim1 << 16));     // dim0 hi | dim1 lo
      g1[3] = (int)((dim1 >> 16) | (dim0 << 16));     // dim1 hi | tile_dim0
      g1[4] = 16;                                     // tile_dim1 = 16 rows
      g1[5] = (int)(unsigned)(st0 & 0xFFFFFFFFu);     // dim0_stride lo
      g1[6] = (int)(unsigned)((st0 >> 32) & 0xFFFFu); // dim0_stride hi
      g1[7] = 0;
      i32x4 z4 = {};
#if __clang_major__ >= 23
      i32x8 z8 = {};
      __builtin_amdgcn_tensor_load_to_lds(g0, g1, z4, z4, z8, 0);
#else
      __builtin_amdgcn_tensor_load_to_lds(g0, g1, z4, z4, 0);
#endif
    }
#endif

    // ---- gh = h @ Whh^T via WMMA, fully branch-free -----------------------
    for (int nt = wave; nt < ntiles; nt += 8) {
      const int n0 = nt * 16;
      const int mloc = lane & 15;
      const int half = lane >> 4;
      const int ncol = n0 + mloc;
      v8f acc = {};
      for (int k0 = 0; k0 < hidPad; k0 += 32) {
        union { v16h v; unsigned u[8]; } ua;
        const unsigned* hp32 = (const unsigned*)(s_h + mloc * GRU_HPAD + k0);
#pragma unroll
        for (int d = 0; d < 4; ++d) {
          ua.u[d]     = hp32[half * 4 + d];
          ua.u[4 + d] = hp32[8 + half * 4 + d];
        }
        union { v16h v; unsigned u[8]; } ub;
        const unsigned* bp =
            (const unsigned*)(Whh + (long long)ncol * hidPad + k0);
#pragma unroll
        for (int d = 0; d < 4; ++d) {
          ub.u[d]     = bp[half * 4 + d];
          ub.u[4 + d] = bp[8 + half * 4 + d];
        }
        acc = __builtin_amdgcn_wmma_f32_16x16x32_f16(false, ua.v, false, ub.v,
                                                     (short)0, acc, false, false);
      }
#pragma unroll
      for (int r = 0; r < 8; ++r)
        s_gh[(half * 8 + r) * GRU_NPAD + ncol] = acc[r];
    }
#ifdef USE_TDM
    if (wave == 0) __builtin_amdgcn_s_wait_tensorcnt(0);
#endif
    __syncthreads();

    // ---- GRU gates (f32 state in registers, f16 mirror in LDS) -----------
#pragma unroll
    for (int i = 0; i < 10; ++i) {
      int e = tid + i * 256;
      if (e < 16 * hid) {
        int lr = e / hid, j = e - lr * hid;
        int gb = b0 + lr;
        if (gb < Bb) {
#ifdef USE_TDM
          float g_r = s_gi[lr * N3pad + j];
          float g_z = s_gi[lr * N3pad + hid + j];
          float g_n = s_gi[lr * N3pad + 2 * hid + j];
#else
          const float* gir = gi + ((long long)gb * T + tg) * N3pad;
          float g_r = gir[j];
          float g_z = gir[hid + j];
          float g_n = gir[2 * hid + j];
#endif
          float hr = s_gh[lr * GRU_NPAD + j]           + bhh[j];
          float hz = s_gh[lr * GRU_NPAD + hid + j]     + bhh[hid + j];
          float hn = s_gh[lr * GRU_NPAD + 2 * hid + j] + bhh[2 * hid + j];
          float r = sigm(g_r + hr);
          float z = sigm(g_z + hz);
          float n = tanhf(g_n + r * hn);
          float hv = (1.0f - z) * n + z * hreg[i];
          hreg[i] = hv;
          s_h[lr * GRU_HPAD + j] = (f16)hv;
          out[((long long)gb * T + tg) * outStride + outOff + j] = hv;
        }
      }
    }
    __syncthreads();
  }
}

// ---------------------------------------------------------------------------
// Row softmax.  Row base = ((r/rpb)*qs + r%rpb)*ld when rpb>0, else r*ld.
// Writes explicit zeros into pad columns [cols, ld).
// ---------------------------------------------------------------------------
__global__ __launch_bounds__(256) void k_softmax(
    const float* __restrict__ in, float* __restrict__ out,
    f16* __restrict__ out16, int cols, int ld, int rpb, int qs)
{
  __shared__ float red[8];
  const int r = blockIdx.x;
  const long long base =
      (rpb > 0) ? ((long long)(r / rpb) * qs + (r % rpb)) * ld : (long long)r * ld;
  const float* x = in + base;
  const int tid = threadIdx.x, lane = tid & 31, wave = tid >> 5;

  float m = -3.4e38f;
  for (int c = tid; c < cols; c += 256) m = fmaxf(m, x[c]);
  m = wave_max(m);
  if (lane == 0) red[wave] = m;
  __syncthreads();
  m = red[0];
#pragma unroll
  for (int w = 1; w < 8; ++w) m = fmaxf(m, red[w]);

  float s = 0.0f;
  for (int c = tid; c < cols; c += 256) s += __expf(x[c] - m);
  s = wave_sum(s);
  __syncthreads();
  if (lane == 0) red[wave] = s;
  __syncthreads();
  s = 0.0f;
#pragma unroll
  for (int w = 0; w < 8; ++w) s += red[w];
  float inv = 1.0f / s;
  for (int c = tid; c < cols; c += 256) {
    float v = __expf(x[c] - m) * inv;
    out[base + c] = v;
    if (out16) out16[base + c] = (f16)v;
  }
  for (int c = cols + tid; c < ld; c += 256) {   // zero the pad columns
    out[base + c] = 0.0f;
    if (out16) out16[base + c] = (f16)0.0f;
  }
}

// out[row] = sum_d X[row,d] * w_row[d];  w_row = w + (row/wDiv)*wLd (wDiv=0: shared)
__global__ __launch_bounds__(256) void k_rowdot(
    const float* __restrict__ X, const float* __restrict__ w,
    float* __restrict__ out, int rows, int D, int wDiv, int wLd)
{
  int gw = (int)((blockIdx.x * 256u + threadIdx.x) >> 5);
  int lane = threadIdx.x & 31;
  if (gw >= rows) return;
  const float* x = X + (long long)gw * D;
  const float* wr = wDiv ? (w + (long long)(gw / wDiv) * wLd) : w;
  float s = 0.0f;
  for (int d = lane; d < D; d += 32) s += x[d] * wr[d];
  s = wave_sum(s);
  if (lane == 0) out[gw] = s;
}

// out[g,d] = sum_t w[g,t] * X[g,t,d]   (block per g)
__global__ __launch_bounds__(256) void k_wsum(
    const float* __restrict__ X, const float* __restrict__ w,
    float* __restrict__ out, f16* __restrict__ out16, int T, int D)
{
  const long long g = blockIdx.x;
  for (int d = threadIdx.x; d < D; d += 256) {
    float s = 0.0f;
    for (int t = 0; t < T; ++t) s += w[g * T + t] * X[(g * T + t) * D + d];
    out[g * D + d] = s;
    if (out16) out16[g * D + d] = (f16)s;
  }
}

// out[row] = sum_k v[k] * tanh( W[k,:] . X[row,:] + Add[row/addDiv, k] )
__global__ __launch_bounds__(128) void k_vproj(
    const float* __restrict__ X, const float* __restrict__ W,
    const float* __restrict__ v, const float* __restrict__ Add,
    int addDiv, int D, int K, float* __restrict__ out)
{
  __shared__ float red[4];
  const long long row = blockIdx.x;
  const int tid = threadIdx.x, lane = tid & 31, wave = tid >> 5;
  float t = 0.0f;
  if (tid < K) {
    const float* x = X + row * D;
    const float* wr = W + (long long)tid * D;
    float a = Add ? Add[(long long)(row / addDiv) * K + tid] : 0.0f;
    for (int d = 0; d < D; ++d) a += wr[d] * x[d];
    t = v[tid] * tanhf(a);
  }
  t = wave_sum(t);
  if (lane == 0) red[wave] = t;
  __syncthreads();
  if (tid == 0) out[row] = red[0] + red[1] + red[2] + red[3];
}

// score[(b*qs+q)*ldS + p] = sum_k v[k]*tanh(sA*Aq[b,p,k] + sB*Bq[b,q,k])
__global__ __launch_bounds__(256) void k_att_addscore(
    const float* __restrict__ Aq, const float* __restrict__ Bq,
    const float* __restrict__ v, float sA, float sB,
    int P_, int K, float* __restrict__ sc, int Q_, int ldS, int qs)
{
  __shared__ float s_b[128], s_v[128];
  const int bq = blockIdx.x;
  const int b = bq / Q_, q = bq - b * Q_;
  for (int k = threadIdx.x; k < K; k += 256) {
    s_b[k] = Bq[((long long)b * Q_ + q) * K + k];
    s_v[k] = v[k];
  }
  __syncthreads();
  for (int p = threadIdx.x; p < P_; p += 256) {
    const float* ar = Aq + ((long long)b * P_ + p) * K;
    float s = 0.0f;
    for (int k = 0; k < K; ++k) s += s_v[k] * tanhf(sA * ar[k] + sB * s_b[k]);
    sc[((long long)b * qs + q) * ldS + p] = s;
  }
}

// score[(b*qs+q)*ldS + p] = sum_k vd[k]*tanh(sum_d hp[b,q,d]*hq[b,p,d]*Wd[k,d])
__global__ __launch_bounds__(256) void k_att_prod(
    const float* __restrict__ hp, const float* __restrict__ hq,
    const float* __restrict__ Wd, const float* __restrict__ vd,
    int Q_, int P_, int D, int K, float* __restrict__ sc, int ldS, int qs)
{
  __shared__ float s_hp[256];
  const int bq = blockIdx.x;
  const int b = bq / Q_, q = bq - b * Q_;
  for (int d = threadIdx.x; d < D; d += 256)
    s_hp[d] = hp[((long long)b * Q_ + q) * D + d];
  __syncthreads();
  const int lane = threadIdx.x & 31, wave = threadIdx.x >> 5;
  for (int p = wave; p < P_; p += 8) {
    const float* hqr = hq + ((long long)b * P_ + p) * D;
    float acc = 0.0f;
    for (int kk = 0; kk < K; kk += 32) {
      int k = kk + lane;
      const float* wr = Wd + (long long)k * D;
      float y = 0.0f;
      for (int d = 0; d < D; ++d) y += s_hp[d] * hqr[d] * wr[d];
      acc += vd[k] * tanhf(y);
    }
    acc = wave_sum(acc);
    if (lane == 0) sc[((long long)b * qs + q) * ldS + p] = acc;
  }
}

// gather embedding rows into f16 with zero-padded row width ldp
__global__ void k_gather_pad(const float* __restrict__ emb, const int* __restrict__ idx,
                             int ntok, int E_, int ldp, f16* __restrict__ o16)
{
  long long i = (long long)blockIdx.x * 256 + threadIdx.x;
  long long total = (long long)ntok * ldp;
  if (i < total) {
    int tkn = (int)(i / ldp);
    int e = (int)(i - (long long)tkn * ldp);
    o16[i] = (e < E_) ? (f16)emb[(long long)idx[tkn] * E_ + e] : (f16)0.0f;
  }
}

__global__ void k_cast16(const float* __restrict__ x, f16* __restrict__ y, long long n)
{
  long long i = (long long)blockIdx.x * 256 + threadIdx.x;
  if (i < n) y[i] = (f16)x[i];
}

// zero-padded f32 -> f16 matrix cast: dst[rowsPad x colsPad] from src[rows x cols]
__global__ void k_cast_pad(const float* __restrict__ src, f16* __restrict__ dst,
                           int rows, int cols, int rowsPad, int colsPad)
{
  long long i = (long long)blockIdx.x * 256 + threadIdx.x;
  long long tot = (long long)rowsPad * colsPad;
  if (i < tot) {
    int r = (int)(i / colsPad), c = (int)(i - (long long)r * colsPad);
    dst[i] = (r < rows && c < cols) ? (f16)src[(long long)r * cols + c] : (f16)0.0f;
  }
}

// batched f32 [G, R, C] -> f16 transpose [G, C, Rpad] with zero pad rows
__global__ void k_transpose16(const float* __restrict__ src, f16* __restrict__ dst,
                              int R, int C, int Rpad)
{
  long long i = (long long)blockIdx.x * 256 + threadIdx.x;
  long long perG = (long long)C * Rpad;
  long long g = i / perG;
  long long rem = i - g * perG;
  int c = (int)(rem / Rpad);
  int r = (int)(rem - (long long)c * Rpad);
  dst[i] = (r < R) ? (f16)src[(g * R + r) * C + c] : (f16)0.0f;
}

// zero-padded f32 vector copy
__global__ void k_padvec(const float* __restrict__ src, float* __restrict__ dst,
                         int n, int npad)
{
  int i = blockIdx.x * 256 + threadIdx.x;
  if (i < npad) dst[i] = (i < n) ? src[i] : 0.0f;
}

// xcat = [a | b]; a read through padded layout (aRpb rows per batch, stride aQs)
__global__ void k_concat2(const float* __restrict__ a, const float* __restrict__ b,
                          int rows, int D, int aRpb, int aQs,
                          float* __restrict__ o32, f16* __restrict__ o16)
{
  long long i = (long long)blockIdx.x * 256 + threadIdx.x;
  long long tot = (long long)rows * 2 * D;
  if (i < tot) {
    int r = (int)(i / (2 * D));
    int c = (int)(i - (long long)r * 2 * D);
    long long ar = (aRpb > 0) ? ((long long)(r / aRpb) * aQs + (r % aRpb)) : r;
    float v = (c < D) ? a[ar * D + c] : b[(long long)r * D + (c - D)];
    o32[i] = v;
    o16[i] = (f16)v;
  }
}

__global__ void k_mulcast(const float* __restrict__ g, const float* __restrict__ x,
                          f16* __restrict__ y, long long n)
{
  long long i = (long long)blockIdx.x * 256 + threadIdx.x;
  if (i < n) y[i] = (f16)(g[i] * x[i]);
}

__global__ void k_stack4(const float* __restrict__ h0, const float* __restrict__ h1,
                         const float* __restrict__ h2, const float* __restrict__ h3,
                         int rows, int D, float* __restrict__ agg)
{
  long long i = (long long)blockIdx.x * 256 + threadIdx.x;
  long long tot = (long long)rows * 4 * D;
  if (i < tot) {
    int d = (int)(i % D);
    long long r4 = i / D;
    int which = (int)(r4 & 3);
    long long r = r4 >> 2;
    const float* src = which == 0 ? h0 : which == 1 ? h1 : which == 2 ? h2 : h3;
    agg[i] = src[r * D + d];
  }
}

// loss = -mean_b log softmax(logits[b,:])[0];  one wave, B=32
__global__ __launch_bounds__(32) void k_loss(const float* __restrict__ logits,
                                             float* __restrict__ out, int Bn)
{
  int tid = threadIdx.x;
  float l = 0.0f;
  if (tid < Bn) {
    float a = logits[tid * 3], b = logits[tid * 3 + 1], c = logits[tid * 3 + 2];
    float m = fmaxf(a, fmaxf(b, c));
    float s = __expf(a - m) + __expf(b - m) + __expf(c - m);
    l = -(a - m - __logf(s));
  }
  l = wave_sum(l);
  if (tid == 0) out[0] = l / (float)Bn;
}

// ---------------------------------------------------------------------------
// Host orchestration
// ---------------------------------------------------------------------------
extern "C" void kernel_launch(void* const* d_in, const int* in_sizes, int n_in,
                              void* d_out, int out_size, void* d_ws, size_t ws_size,
                              hipStream_t stream)
{
  (void)in_sizes; (void)n_in; (void)out_size; (void)ws_size;
  const int Bn = 32, Qn = 50, Pn = 300, An = 20, En = 300, Hn = 128, HIDA = 150;
  const int D2 = 256, D4 = 512;
  const int EnP = 320;     // E padded to K%32==0
  const int QP = 64;       // per-batch Q rows padded to 16
  const int PP = 320;      // per-batch P cols padded (K and N pads)
  const int EncP = 304;    // Wpred output N pad

  static const int AENCG[8] = {7, 8, 9, 10, 11, 12, 13, 14};
  static const int QENCG[8] = {16, 17, 18, 19, 20, 21, 22, 23};
  static const int PENCG[8] = {24, 25, 26, 27, 28, 29, 30, 31};
  static const int HTCG[8]  = {41, 42, 43, 44, 45, 46, 47, 48};
  static const int HTBG[8]  = {50, 51, 52, 53, 54, 55, 56, 57};
  static const int HTDG[8]  = {59, 60, 61, 62, 63, 64, 65, 66};
  static const int HTMG[8]  = {68, 69, 70, 71, 72, 73, 74, 75};
  static const int AGGG[8]  = {76, 77, 78, 79, 80, 81, 82, 83};

  auto F32 = [&](int i) { return (const float*)d_in[i]; };
  auto I32 = [&](int i) { return (const int*)d_in[i]; };

  char* base = (char*)d_ws;
  size_t off = 0;
  auto alloc = [&](size_t bytes) -> void* {
    size_t a = (off + 255) & ~(size_t)255;
    off = a + bytes;
    return (void*)(base + a);
  };
  auto allocF = [&](long long n) { return (float*)alloc((size_t)n * 4); };
  auto allocH = [&](long long n) { return (f16*)alloc((size_t)n * 2); };

  auto cast16 = [&](const float* src, long long n, long long slack) -> f16* {
    f16* p = allocH(n + slack);
    k_cast16<<<dim3((unsigned)((n + 255) / 256)), 256, 0, stream>>>(src, p, n);
    return p;
  };
  auto wcast = [&](int idx, long long n) { return cast16(F32(idx), n, 0); };
  auto padcast = [&](int idx, int rows, int cols, int rowsPad, int colsPad) -> f16* {
    long long tot = (long long)rowsPad * colsPad;
    f16* p = allocH(tot);
    k_cast_pad<<<dim3((unsigned)((tot + 255) / 256)), 256, 0, stream>>>(
        F32(idx), p, rows, cols, rowsPad, colsPad);
    return p;
  };
  auto padvec = [&](int idx, int n, int npad) -> float* {
    float* p = allocF(npad);
    k_padvec<<<dim3((unsigned)((npad + 255) / 256)), 256, 0, stream>>>(
        F32(idx), p, n, npad);
    return p;
  };

  // all M,N,K here are padded: M%16==0, N%16==0, K%32==0; all operands K-major
  auto gemm = [&](const f16* A, const f16* Bw, const float* bias, float* C, f16* C16,
                  int M, int N, int K, int lda, int ldb, int ldc, int act,
                  int batch = 1, long long sA = 0, long long sB = 0, long long sC = 0) {
    dim3 g((unsigned)(N / 16), (unsigned)(M / 16), (unsigned)batch);
    k_gemm<<<g, 32, 0, stream>>>(A, Bw, bias, C, C16, K, lda, ldb, ldc,
                                 act, sA, sB, sC);
  };

  // shared gi scratch, sized for the largest padded projection (9600 x 384)
  float* gi = allocF((long long)Bn * Pn * 384);

  auto bigru = [&](const f16* x16, int Bb, int T, int inD, int inDpad, int hid,
                   const int* gx, float* outp) {
    int N3 = 3 * hid;
    int hidPad = (hid + 31) & ~31;
    int N3pad = (N3 + 15) & ~15;
    f16* wihF = padcast(gx[0], N3, inD, N3pad, inDpad);
    f16* whhF = padcast(gx[1], N3, hid, N3pad, hidPad);
    float* bihF = padvec(gx[2], N3, N3pad);
    f16* wihB = padcast(gx[4], N3, inD, N3pad, inDpad);
    f16* whhB = padcast(gx[5], N3, hid, N3pad, hidPad);
    float* bihB = padvec(gx[6], N3, N3pad);
    gemm(x16, wihF, bihF, gi, nullptr, Bb * T, N3pad, inDpad, inDpad, inDpad, N3pad, 0);
    k_gru_dir<<<dim3((unsigned)((Bb + 15) / 16)), 256, 0, stream>>>(
        gi, whhF, F32(gx[3]), outp, Bb, T, hid, hidPad, N3pad, 0, 2 * hid, 0);
    gemm(x16, wihB, bihB, gi, nullptr, Bb * T, N3pad, inDpad, inDpad, inDpad, N3pad, 0);
    k_gru_dir<<<dim3((unsigned)((Bb + 15) / 16)), 256, 0, stream>>>(
        gi, whhB, F32(gx[7]), outp, Bb, T, hid, hidPad, N3pad, 1, 2 * hid, hid);
  };

  // ---- answer branch: embed -> bigru(hid=150) -> attention pooling -------
  f16* ae16 = allocH((long long)Bn * 3 * An * EnP);
  k_gather_pad<<<dim3((unsigned)(((long long)Bn * 3 * An * EnP + 255) / 256)), 256, 0, stream>>>(
      F32(6), I32(2), Bn * 3 * An, En, EnP, ae16);
  float* a_enc = allocF((long long)Bn * 3 * An * 2 * HIDA);
  bigru(ae16, Bn * 3, An, En, EnP, HIDA, AENCG, a_enc);
  float* a_sc = allocF((long long)Bn * 3 * An);
  k_rowdot<<<dim3((unsigned)((Bn * 3 * An + 7) / 8)), 256, 0, stream>>>(
      a_enc, F32(15), a_sc, Bn * 3 * An, 2 * HIDA, 0, 0);
  float* a_soft = allocF((long long)Bn * 3 * An);
  k_softmax<<<dim3((unsigned)(Bn * 3)), 256, 0, stream>>>(a_sc, a_soft, nullptr,
                                                          An, An, 0, 0);
  float* a_emb = allocF((long long)Bn * 3 * En);
  k_wsum<<<dim3((unsigned)(Bn * 3)), 256, 0, stream>>>(a_enc, a_soft, a_emb, nullptr,
                                                       An, 2 * HIDA);

  // ---- encoders (note source swap: hq=passage, hp=question) --------------
  f16* pe16 = allocH((long long)Bn * Pn * EnP);
  k_gather_pad<<<dim3((unsigned)(((long long)Bn * Pn * EnP + 255) / 256)), 256, 0, stream>>>(
      F32(6), I32(1), Bn * Pn, En, EnP, pe16);
  f16* qe16 = allocH((long long)Bn * Qn * EnP);
  k_gather_pad<<<dim3((unsigned)(((long long)Bn * Qn * EnP + 255) / 256)), 256, 0, stream>>>(
      F32(6), I32(0), Bn * Qn, En, EnP, qe16);
  float* hq = allocF((long long)Bn * Pn * D2);
  bigru(pe16, Bn, Pn, En, EnP, Hn, QENCG, hq);
  f16* hq16 = cast16(hq, (long long)Bn * Pn * D2, 8192);
  // transposed hq for pooling: [B, D2, PP] f16, zero-padded P columns
  f16* hqT16 = allocH((long long)Bn * D2 * PP);
  k_transpose16<<<dim3((unsigned)(((long long)Bn * D2 * PP + 255) / 256)), 256, 0, stream>>>(
      hq, hqT16, Pn, D2, PP);
  float* hp = allocF((long long)Bn * Qn * D2);
  bigru(qe16, Bn, Qn, En, EnP, Hn, PENCG, hp);
  f16* hp16 = cast16(hp, (long long)Bn * Qn * D2, 8192);      // +slack for M spill

  // ---- attention precomputes ---------------------------------------------
  f16* Wc1h = wcast(32, (long long)Hn * D2);
  f16* Wc2h = wcast(33, (long long)Hn * D2);
  f16* Wbh  = wcast(35, (long long)D2 * D2);
  f16* Wmh  = wcast(38, (long long)Hn * D2);
  float* hqWc1 = allocF((long long)Bn * Pn * Hn);
  gemm(hq16, Wc1h, nullptr, hqWc1, nullptr, Bn * Pn, Hn, D2, D2, D2, Hn, 0);
  float* hpWc2 = allocF((long long)Bn * Qn * Hn);
  gemm(hp16, Wc2h, nullptr, hpWc2, nullptr, Bn * Qn, Hn, D2, D2, D2, Hn, 0);
  float* hqWb = allocF((long long)Bn * Pn * D2);
  f16* hqWb16 = allocH((long long)Bn * Pn * D2 + 8192);        // +slack for N spill
  gemm(hq16, Wbh, nullptr, hqWb, hqWb16, Bn * Pn, D2, D2, D2, D2, D2, 0);
  float* hqWm = allocF((long long)Bn * Pn * Hn);
  gemm(hq16, Wmh, nullptr, hqWm, nullptr, Bn * Pn, Hn, D2, D2, D2, Hn, 0);
  float* hpWm = allocF((long long)Bn * Qn * Hn);
  gemm(hp16, Wmh, nullptr, hpWm, nullptr, Bn * Qn, Hn, D2, D2, D2, Hn, 0);

  // ---- four attention flavors (padded score layout [B, QP, PP]) ----------
  float* score = allocF((long long)Bn * QP * PP);
  float* soft  = allocF((long long)Bn * QP * PP);
  f16* soft16  = allocH((long long)Bn * QP * PP);
  float* qt[4];
  for (int i = 0; i < 4; ++i) qt[i] = allocF((long long)Bn * QP * D2);

  auto pool = [&](float* dst) {
    k_softmax<<<dim3((unsigned)(Bn * Qn)), 256, 0, stream>>>(score, soft, soft16,
                                                             Pn, PP, Qn, QP);
    gemm(soft16, hqT16, nullptr, dst, nullptr, QP, D2, PP, PP, PP, D2, 0,
         Bn, (long long)QP * PP, (long long)D2 * PP, (long long)QP * D2);
  };

  k_att_addscore<<<dim3((unsigned)(Bn * Qn)), 256, 0, stream>>>(
      hqWc1, hpWc2, F32(34), 1.0f, 1.0f, Pn, Hn, score, Qn, PP, QP);
  pool(qt[0]);
  gemm(hp16, hqWb16, nullptr, score, nullptr, QP, PP, D2, D2, D2, PP, 0,
       Bn, (long long)Qn * D2, (long long)Pn * D2, (long long)QP * PP);
  pool(qt[1]);
  k_att_prod<<<dim3((unsigned)(Bn * Qn)), 256, 0, stream>>>(
      hp, hq, F32(36), F32(37), Qn, Pn, D2, Hn, score, PP, QP);
  pool(qt[2]);
  k_att_addscore<<<dim3((unsigned)(Bn * Qn)), 256, 0, stream>>>(
      hqWm, hpWm, F32(39), 1.0f, -1.0f, Pn, Hn, score, Qn, PP, QP);
  pool(qt[3]);

  // ---- gated bigru heads --------------------------------------------------
  float* xcat = allocF((long long)Bn * Qn * D4);
  f16* xcat16 = allocH((long long)Bn * Qn * D4);
  float* gbuf = allocF((long long)Bn * Qn * D4);
  f16* xg16 = allocH((long long)Bn * Qn * D4);
  float* ht[4];
  const int WGIDX[4] = {40, 49, 58, 67};
  const int* GRUX[4] = {HTCG, HTBG, HTDG, HTMG};
  for (int i = 0; i < 4; ++i) {
    ht[i] = allocF((long long)Bn * Qn * D2);
    k_concat2<<<dim3((unsigned)(((long long)Bn * Qn * D4 + 255) / 256)), 256, 0, stream>>>(
        qt[i], hp, Bn * Qn, D2, Qn, QP, xcat, xcat16);
    f16* Wg16 = wcast(WGIDX[i], (long long)D4 * D4);
    gemm(xcat16, Wg16, nullptr, gbuf, nullptr, Bn * Qn, D4, D4, D4, D4, D4, 1);
    k_mulcast<<<dim3((unsigned)(((long long)Bn * Qn * D4 + 255) / 256)), 256, 0, stream>>>(
        gbuf, xcat, xg16, (long long)Bn * Qn * D4);
    bigru(xg16, Bn, Qn, D4, D4, Hn, GRUX[i], ht[i]);
  }

  // ---- aggregation ---------------------------------------------------------
  float* agg = allocF((long long)Bn * Qn * 4 * D2);
  k_stack4<<<dim3((unsigned)(((long long)Bn * Qn * 4 * D2 + 255) / 256)), 256, 0, stream>>>(
      ht[0], ht[1], ht[2], ht[3], Bn * Qn, D2, agg);
  float* sagg = allocF((long long)Bn * Qn * 4);
  k_vproj<<<dim3((unsigned)(Bn * Qn * 4)), 128, 0, stream>>>(
      agg, F32(84), F32(85), nullptr, 1, D2, Hn, sagg);
  float* wagg = allocF((long long)Bn * Qn * 4);
  k_softmax<<<dim3((unsigned)(Bn * Qn)), 256, 0, stream>>>(sagg, wagg, nullptr, 4, 4, 0, 0);
  float* rep = allocF((long long)Bn * Qn * D2);
  f16* rep16 = allocH((long long)Bn * Qn * D2);
  k_wsum<<<dim3((unsigned)(Bn * Qn)), 256, 0, stream>>>(agg, wagg, rep, rep16, 4, D2);
  float* agg_rep = allocF((long long)Bn * Qn * D2);
  bigru(rep16, Bn, Qn, D2, D2, Hn, AGGG, agg_rep);

  // ---- prediction head -----------------------------------------------------
  float* sjq = allocF((long long)Bn * Pn);
  k_vproj<<<dim3((unsigned)(Bn * Pn)), 128, 0, stream>>>(
      hq, F32(86), F32(87), nullptr, 1, D2, Hn, sjq);
  float* wq = allocF((long long)Bn * Pn);
  k_softmax<<<dim3((unsigned)Bn), 256, 0, stream>>>(sjq, wq, nullptr, Pn, Pn, 0, 0);
  float* rq = allocF((long long)Bn * D2);
  f16* rq16 = allocH((long long)Bn * D2);
  k_wsum<<<dim3((unsigned)Bn), 256, 0, stream>>>(hq, wq, rq, rq16, Pn, D2);
  f16* Wp2h = wcast(89, (long long)Hn * D2);
  float* rqW = allocF((long long)Bn * Hn);
  gemm(rq16, Wp2h, nullptr, rqW, nullptr, Bn, Hn, D2, D2, D2, Hn, 0);
  float* sjp = allocF((long long)Bn * Qn);
  k_vproj<<<dim3((unsigned)(Bn * Qn)), 128, 0, stream>>>(
      agg_rep, F32(88), F32(90), rqW, Qn, D2, Hn, sjp);
  float* wp = allocF((long long)Bn * Qn);
  k_softmax<<<dim3((unsigned)Bn), 256, 0, stream>>>(sjp, wp, nullptr, Qn, Qn, 0, 0);
  float* rp = allocF((long long)Bn * D2);
  f16* rp16 = allocH((long long)Bn * D2);
  k_wsum<<<dim3((unsigned)Bn), 256, 0, stream>>>(agg_rep, wp, rp, rp16, Qn, D2);
  f16* Wpredh = padcast(91, En, D2, EncP, D2);
  float* encb = allocF((long long)Bn * EncP);
  gemm(rp16, Wpredh, nullptr, encb, nullptr, Bn, EncP, D2, D2, D2, EncP, 3);
  float* logits = allocF((long long)Bn * 3);
  k_rowdot<<<dim3((unsigned)((Bn * 3 + 7) / 8)), 256, 0, stream>>>(
      a_emb, encb, logits, Bn * 3, En, 3, EncP);
  k_loss<<<1, 32, 0, stream>>>(logits, (float*)d_out, Bn);
}
